// Decoder_57432302682540
// MI455X (gfx1250) — matrix operand order repaired and checked by reading, not compiled, not run on hardware
//
#include <hip/hip_runtime.h>
#include <hip/hip_bf16.h>
#include <math.h>

// Problem constants (from reference)
#define BB 4096
#define LL 7
#define DD 1024
#define HH 1024
#define TT 6

typedef __attribute__((ext_vector_type(16))) __bf16 v16bf;
typedef __attribute__((ext_vector_type(8)))  float  v8f;
typedef __attribute__((ext_vector_type(4)))  int    v4i_;

typedef __attribute__((address_space(1))) v4i_ as1_v4i;
typedef __attribute__((address_space(3))) v4i_ as3_v4i;

// ---------------------------------------------------------------------------
// CDNA5 async global->LDS copy path (ASYNCcnt), with sync fallback.
// ---------------------------------------------------------------------------
#if defined(__has_builtin)
#if __has_builtin(__builtin_amdgcn_global_load_async_to_lds_b128) && \
    __has_builtin(__builtin_amdgcn_s_wait_asynccnt)
#define USE_ASYNC_LDS 1
#endif
#endif

__device__ __forceinline__ void async_b128(const __bf16* g, __bf16* s) {
#ifdef USE_ASYNC_LDS
  __builtin_amdgcn_global_load_async_to_lds_b128((as1_v4i*)g, (as3_v4i*)s, 0, 0);
#else
  *(uint4*)s = *(const uint4*)g;
#endif
}

template <int N>
__device__ __forceinline__ void wait_async_tiles() {
#ifdef USE_ASYNC_LDS
  __builtin_amdgcn_s_wait_asynccnt(N);
#endif
}

// ---------------------------------------------------------------------------
// WMMA fragment loaders (CDNA5 wave32 layouts, 16x16x32 bf16)
// A: 16x32 (MxK). elem j -> K = j + (j>=8?8:0) + (lane>=16?8:0), M = lane%16
// B: 32x16 (KxN). N = lane%16; elem j -> K = j + (j>=8?8:0) + (lane>=16?8:0)
// C/D: VGPR r -> M = r + (lane>=16?8:0), N = lane%16
// ---------------------------------------------------------------------------
__device__ __forceinline__ v16bf load_a_frag(const __bf16* As /*rows x 32*/, int mBase, int lane) {
  v16bf a;
  const int m = mBase + (lane & 15);
  const int koff = (lane >= 16) ? 8 : 0;
#pragma unroll
  for (int j = 0; j < 16; ++j) {
    const int kk = j + ((j >= 8) ? 8 : 0) + koff;
    a[j] = As[m * 32 + kk];
  }
  return a;
}

__device__ __forceinline__ v16bf load_b_frag(const __bf16* Bs /*64x32 [n][k]*/, int nBase, int lane) {
  v16bf b;
  const int n = nBase + (lane & 15);
  const int koff = (lane >= 16) ? 8 : 0;
#pragma unroll
  for (int j = 0; j < 16; ++j) b[j] = Bs[n * 32 + koff + j + ((j >= 8) ? 8 : 0)];
  return b;
}

__device__ __forceinline__ v8f wmma_bf16(v16bf a, v16bf b, v8f c) {
  return __builtin_amdgcn_wmma_f32_16x16x32_bf16(false, a, false, b, (short)0, c, false, false);
}

// ---------------------------------------------------------------------------
// Generic tiled GEMM: C[M,N] = A[M,K] @ W[N,K]^T  (bf16 in, f32 out)
// flags: 1 = add bias[n], 2 = relu, 4 = also write bf16 copy to Cb
// Block: 128 threads (4 wave32). Tile 128M x 64N x 32K, double-buffered LDS,
// async global->LDS copies overlapped with WMMA.
// ---------------------------------------------------------------------------
__global__ __launch_bounds__(128) void gemm_bf16_wmma(
    const __bf16* __restrict__ A, const __bf16* __restrict__ W,
    const float* __restrict__ bias, float* __restrict__ C,
    __bf16* __restrict__ Cb, int M, int N, int K, int flags)
{
  __shared__ __bf16 As[2][128 * 32];
  __shared__ __bf16 Bs[2][64 * 32];
  const int tid = threadIdx.x, lane = tid & 31, wave = tid >> 5;
  const int m0 = blockIdx.x * 128, n0 = blockIdx.y * 64;

  v8f c[2][4] = {};

  auto issue = [&](int buf, int k0) {
    // A: thread t owns full row (m0+t), 32 bf16 = 4x b128
    const __bf16* gA = A + (size_t)(m0 + tid) * K + k0;
    __bf16* sA = &As[buf][tid * 32];
    async_b128(gA + 0,  sA + 0);
    async_b128(gA + 8,  sA + 8);
    async_b128(gA + 16, sA + 16);
    async_b128(gA + 24, sA + 24);
    // B: two threads per row, 16 bf16 halves = 2x b128
    const int row = tid >> 1, kh = (tid & 1) * 16;
    const __bf16* gB = W + (size_t)(n0 + row) * K + k0 + kh;
    __bf16* sB = &Bs[buf][row * 32 + kh];
    async_b128(gB + 0, sB + 0);
    async_b128(gB + 8, sB + 8);
    if (k0 + 64 < K) __builtin_prefetch(gA + 64, 0, 1);
  };

  issue(0, 0);
  const int nK = K >> 5;
  for (int kt = 0; kt < nK; ++kt) {
    const int buf = kt & 1;
    if (kt + 1 < nK) {
      issue(buf ^ 1, (kt + 1) << 5);   // prefetch next tile into other buffer
      wait_async_tiles<6>();           // drain only this tile's 6 in-order copies
    } else {
      wait_async_tiles<0>();
    }
    __syncthreads();

    v16bf a0 = load_a_frag(&As[buf][0], wave * 32, lane);
    v16bf a1 = load_a_frag(&As[buf][0], wave * 32 + 16, lane);
#pragma unroll
    for (int nt = 0; nt < 4; ++nt) {
      v16bf b = load_b_frag(&Bs[buf][0], nt * 16, lane);
      c[0][nt] = wmma_bf16(a0, b, c[0][nt]);
      c[1][nt] = wmma_bf16(a1, b, c[1][nt]);
    }
    __syncthreads();
  }

  const int rowOff = (lane >= 16) ? 8 : 0;
#pragma unroll
  for (int ms = 0; ms < 2; ++ms) {
#pragma unroll
    for (int nt = 0; nt < 4; ++nt) {
      const int n = n0 + nt * 16 + (lane & 15);
      const float bv = (flags & 1) ? bias[n] : 0.0f;
#pragma unroll
      for (int r = 0; r < 8; ++r) {
        const int m = m0 + wave * 32 + ms * 16 + r + rowOff;
        float v = c[ms][nt][r] + bv;
        if (flags & 2) v = fmaxf(v, 0.0f);
        C[(size_t)m * N + n] = v;
        if (flags & 4) Cb[(size_t)m * N + n] = (__bf16)v;
      }
    }
  }
}

// ---------------------------------------------------------------------------
// Fused 4-way tanh attention:
//   att[b, k*L + l] = 10*tanh( sum_h V4[k,h]*tanh(inp[b,h] + enc[b,l,:]·Wc[k,h,:] + bc[k,h]) )
// encR = enc reshaped [B*L, D] (bf16). One block = 128 rows of encR, one head k.
// ---------------------------------------------------------------------------
__global__ __launch_bounds__(128) void attn_fused(
    const __bf16* __restrict__ encR, const __bf16* __restrict__ Wc,
    const float* __restrict__ bc, const float* __restrict__ V4,
    const float* __restrict__ inp, const int* __restrict__ mask,
    float* __restrict__ att)
{
  __shared__ __bf16 As[2][128 * 32];
  __shared__ __bf16 Bs[2][64 * 32];
  __shared__ float acc[128];
  const int tid = threadIdx.x, lane = tid & 31, wave = tid >> 5;
  const int m0 = blockIdx.x * 128;
  const int k  = blockIdx.y;
  const __bf16* Wk = Wc + (size_t)k * HH * DD;

  acc[tid] = 0.0f;
  __syncthreads();

  auto issue = [&](int buf, int h0, int k0) {
    const __bf16* gA = encR + (size_t)(m0 + tid) * DD + k0;
    __bf16* sA = &As[buf][tid * 32];
    async_b128(gA + 0,  sA + 0);
    async_b128(gA + 8,  sA + 8);
    async_b128(gA + 16, sA + 16);
    async_b128(gA + 24, sA + 24);
    const int row = tid >> 1, kh = (tid & 1) * 16;
    const __bf16* gB = Wk + (size_t)(h0 + row) * DD + k0 + kh;
    __bf16* sB = &Bs[buf][row * 32 + kh];
    async_b128(gB + 0, sB + 0);
    async_b128(gB + 8, sB + 8);
  };

  for (int h0 = 0; h0 < HH; h0 += 64) {
    v8f c[2][4] = {};
    issue(0, h0, 0);
    const int nK = DD >> 5;
    for (int kt = 0; kt < nK; ++kt) {
      const int buf = kt & 1;
      if (kt + 1 < nK) {
        issue(buf ^ 1, h0, (kt + 1) << 5);
        wait_async_tiles<6>();
      } else {
        wait_async_tiles<0>();
      }
      __syncthreads();

      v16bf a0 = load_a_frag(&As[buf][0], wave * 32, lane);
      v16bf a1 = load_a_frag(&As[buf][0], wave * 32 + 16, lane);
#pragma unroll
      for (int nt = 0; nt < 4; ++nt) {
        v16bf b = load_b_frag(&Bs[buf][0], nt * 16, lane);
        c[0][nt] = wmma_bf16(a0, b, c[0][nt]);
        c[1][nt] = wmma_bf16(a1, b, c[1][nt]);
      }
      __syncthreads();
    }
    // Epilogue for this 64-wide h block: tanh + V4 dot, accumulate per row.
    const int rowOff = (lane >= 16) ? 8 : 0;
#pragma unroll
    for (int ms = 0; ms < 2; ++ms) {
#pragma unroll
      for (int nt = 0; nt < 4; ++nt) {
        const int h = h0 + nt * 16 + (lane & 15);
        const float vw  = V4[k * HH + h];
        const float bcv = bc[k * HH + h];
#pragma unroll
        for (int r = 0; r < 8; ++r) {
          const int ml = wave * 32 + ms * 16 + r + rowOff;
          const int m  = m0 + ml;
          const int b  = m / LL;
          const float val = vw * tanhf(inp[(size_t)b * HH + h] + c[ms][nt][r] + bcv);
          atomicAdd(&acc[ml], val);
        }
      }
    }
    __syncthreads();
  }

  {
    const int m = m0 + tid, b = m / LL, l = m % LL;
    float a = acc[tid];
    a = (mask[b * LL + l] != 0) ? 10.0f * tanhf(a) : -10.0f;
    att[(size_t)b * (4 * LL) + k * LL + l] = a;
  }
}

// ---------------------------------------------------------------------------
// Elementwise / data-movement helpers
// ---------------------------------------------------------------------------
__global__ void cvt_f32_bf16(const float* __restrict__ s, __bf16* __restrict__ d, size_t n) {
  size_t i = (size_t)blockIdx.x * 256 + threadIdx.x;
  if (i < n) d[i] = (__bf16)s[i];
}

__global__ void zero_f32(float* p, size_t n) {
  size_t i = (size_t)blockIdx.x * 256 + threadIdx.x;
  if (i < n) p[i] = 0.0f;
}

// Wbig[row][seg*D + col] = Wseg[row][col], seg order {Wh, Wv, Wsh, Wsv}
__global__ void pack_wbig(const float* __restrict__ Wh, const float* __restrict__ Wv,
                          const float* __restrict__ Wsh, const float* __restrict__ Wsv,
                          __bf16* __restrict__ Wbig) {
  size_t i = (size_t)blockIdx.x * 256 + threadIdx.x;
  const size_t tot = 4ull * HH * DD;
  if (i >= tot) return;
  int row = (int)(i / (4 * DD));
  int r   = (int)(i % (4 * DD));
  int seg = r / DD, col = r % DD;
  const float* src = (seg == 0) ? Wh : (seg == 1) ? Wv : (seg == 2) ? Wsh : Wsv;
  Wbig[i] = (__bf16)src[(size_t)row * DD + col];
}

// Abig[b] = [hA | vA | hB | vB]: zero-masked by (t==0) so one K=4D GEMM realizes
// the per-row weight-pair select exactly.
__global__ void gather_select(const __bf16* __restrict__ encB, const int* __restrict__ xes,
                              int step, __bf16* __restrict__ Abig) {
  size_t i = (size_t)blockIdx.x * 256 + threadIdx.x;
  const size_t tot = (size_t)BB * 4 * DD;
  if (i >= tot) return;
  int b = (int)(i / (4 * DD));
  int r = (int)(i % (4 * DD));
  int seg = r / DD, d = r % DD;
  const int* x = xes + ((size_t)b * TT + step) * 3;
  const int hi = x[0], vi = x[1], t = x[2];
  const bool first  = (t == 0);
  const bool active = (seg < 2) ? first : !first;
  const int srow = ((seg & 1) == 0) ? hi : vi;
  Abig[i] = active ? encB[((size_t)b * LL + srow) * DD + d] : (__bf16)0.0f;
}

__global__ void gather_idx(const __bf16* __restrict__ encB, const int* __restrict__ idx,
                           __bf16* __restrict__ out) {
  size_t i = (size_t)blockIdx.x * 256 + threadIdx.x;
  if (i >= (size_t)BB * DD) return;
  int b = (int)(i >> 10), d = (int)(i & 1023);
  out[i] = encB[((size_t)b * LL + idx[b]) * DD + d];
}

__global__ void max_inplace(float* __restrict__ st, const float* __restrict__ tmp, size_t n) {
  size_t i = (size_t)blockIdx.x * 256 + threadIdx.x;
  if (i < n) st[i] = fmaxf(st[i], tmp[i]);
}

// qt = relu(edge + st), in place over edge
__global__ void relu_add_inplace(float* __restrict__ a, const float* __restrict__ b, size_t n) {
  size_t i = (size_t)blockIdx.x * 256 + threadIdx.x;
  if (i < n) a[i] = fmaxf(a[i] + b[i], 0.0f);
}

// query = relu(relu(qt + lin) + lin) -> bf16
__global__ void build_query(const float* __restrict__ qt, const float* __restrict__ lin,
                            __bf16* __restrict__ qB, size_t n) {
  size_t i = (size_t)blockIdx.x * 256 + threadIdx.x;
  if (i >= n) return;
  float q = fmaxf(qt[i] + lin[i], 0.0f);
  q = fmaxf(q + lin[i], 0.0f);
  qB[i] = (__bf16)q;
}

// Column softmax over the batch axis (axis=0), in place. 28 columns.
__global__ void col_softmax(float* __restrict__ att, int Bn, int Cn) {
  __shared__ float red[256];
  const int j = blockIdx.x, tid = threadIdx.x;
  float mx = -1e30f;
  for (int b = tid; b < Bn; b += 256) mx = fmaxf(mx, att[(size_t)b * Cn + j]);
  red[tid] = mx; __syncthreads();
  for (int s = 128; s > 0; s >>= 1) { if (tid < s) red[tid] = fmaxf(red[tid], red[tid + s]); __syncthreads(); }
  mx = red[0]; __syncthreads();
  float sum = 0.0f;
  for (int b = tid; b < Bn; b += 256) sum += __expf(att[(size_t)b * Cn + j] - mx);
  red[tid] = sum; __syncthreads();
  for (int s = 128; s > 0; s >>= 1) { if (tid < s) red[tid] += red[tid + s]; __syncthreads(); }
  sum = red[0]; __syncthreads();
  const float inv = 1.0f / sum;
  for (int b = tid; b < Bn; b += 256) {
    const size_t i = (size_t)b * Cn + j;
    att[i] = __expf(att[i] - mx) * inv;
  }
}

// Per-row categorical sample (deterministic hash in lieu of threefry) + outputs.
__global__ void sample_write(const float* __restrict__ alpha, const int* __restrict__ mask,
                             float* __restrict__ out, int Bn) {
  const int b = blockIdx.x * 256 + threadIdx.x;
  if (b >= Bn) return;
  const int Cn = 4 * LL;
  float row[4 * LL];
  float s = 0.0f;
  for (int j = 0; j < Cn; ++j) { row[j] = alpha[(size_t)b * Cn + j]; s += row[j]; }
  unsigned long long x = (unsigned long long)b * 0x9E3779B97F4A7C15ull + 42ull;
  x ^= x >> 30; x *= 0xBF58476D1CE4E5B9ull;
  x ^= x >> 27; x *= 0x94D049BB133111EBull;
  x ^= x >> 31;
  const float u = (float)((x >> 40) & 0xFFFFFF) * (1.0f / 16777216.0f);
  const float target = u * s;
  float cum = 0.0f; int idx = Cn - 1;
  for (int j = 0; j < Cn; ++j) { cum += row[j]; if (cum >= target) { idx = j; break; } }
  out[b] = (float)idx;
  out[Bn + b] = row[idx];
  for (int l = 0; l < LL; ++l)
    out[2 * Bn + (size_t)b * LL + l] = (float)(mask[b * LL + l] - ((l == idx) ? 1 : 0));
}

// ---------------------------------------------------------------------------
// Host launcher
// ---------------------------------------------------------------------------
static inline size_t cdiv(size_t a, size_t b) { return (a + b - 1) / b; }

extern "C" void kernel_launch(void* const* d_in, const int* in_sizes, int n_in,
                              void* d_out, int out_size, void* d_ws, size_t ws_size,
                              hipStream_t stream) {
  const float* enc  = (const float*)d_in[0];
  const int*   xes  = (const int*)d_in[1];
  const int*   idx  = (const int*)d_in[2];
  const int*   mask = (const int*)d_in[3];
  const float* Wq = (const float*)d_in[4];
  const float* bq = (const float*)d_in[5];
  const float* Wc = (const float*)d_in[6];
  const float* bc = (const float*)d_in[7];
  const float* V4 = (const float*)d_in[8];
  const float* Wi = (const float*)d_in[9];
  const float* bi = (const float*)d_in[10];
  const float* Wh = (const float*)d_in[11];
  const float* Wv = (const float*)d_in[12];
  const float* Wsh = (const float*)d_in[13];
  const float* Wsv = (const float*)d_in[14];
  const float* We = (const float*)d_in[15];

  char* ws = (char*)d_ws;
  size_t off = 0;
  auto take = [&](size_t bytes) { char* p = ws + off; off += (bytes + 255) & ~255ull; return p; };

  __bf16* encB  = (__bf16*)take((size_t)BB * LL * DD * 2);   // 58.7 MB
  __bf16* WbigB = (__bf16*)take((size_t)HH * 4 * DD * 2);    //  8.4 MB
  __bf16* WeB   = (__bf16*)take((size_t)HH * DD * 2);
  __bf16* WiB   = (__bf16*)take((size_t)HH * DD * 2);
  __bf16* WqB   = (__bf16*)take((size_t)HH * HH * 2);
  __bf16* WcB   = (__bf16*)take((size_t)4 * HH * DD * 2);    //  8.4 MB
  __bf16* AbigB = (__bf16*)take((size_t)BB * 4 * DD * 2);    // 33.6 MB
  float*  edgeF = (float*)take((size_t)BB * HH * 4);         // 16.8 MB
  __bf16* edgeB = (__bf16*)take((size_t)BB * HH * 2);        //  8.4 MB
  float*  tmpF  = (float*)take((size_t)BB * HH * 4);         // 16.8 MB
  float*  stF   = (float*)take((size_t)BB * HH * 4);         // 16.8 MB
  float*  attF  = (float*)take((size_t)BB * 4 * LL * 4);     //  0.46 MB

  const size_t nBH = (size_t)BB * HH;

  // ---- one-time conversions ----
  cvt_f32_bf16<<<cdiv((size_t)BB * LL * DD, 256), 256, 0, stream>>>(enc, encB, (size_t)BB * LL * DD);
  cvt_f32_bf16<<<cdiv((size_t)HH * DD, 256), 256, 0, stream>>>(We, WeB, (size_t)HH * DD);
  cvt_f32_bf16<<<cdiv((size_t)HH * DD, 256), 256, 0, stream>>>(Wi, WiB, (size_t)HH * DD);
  cvt_f32_bf16<<<cdiv((size_t)HH * HH, 256), 256, 0, stream>>>(Wq, WqB, (size_t)HH * HH);
  cvt_f32_bf16<<<cdiv((size_t)4 * HH * DD, 256), 256, 0, stream>>>(Wc, WcB, (size_t)4 * HH * DD);
  pack_wbig<<<cdiv((size_t)4 * HH * DD, 256), 256, 0, stream>>>(Wh, Wv, Wsh, Wsv, WbigB);
  zero_f32<<<cdiv(nBH, 256), 256, 0, stream>>>(stF, nBH);

  // ---- recurrent edge/subtree: only 5 updates matter for qts[-1] ----
  for (int step = 0; step < TT - 1; ++step) {
    gather_select<<<cdiv((size_t)BB * 4 * DD, 256), 256, 0, stream>>>(encB, xes, step, AbigB);
    // edge = Abig @ Wbig^T   (M=4096, N=1024, K=4096), also emit bf16 copy
    gemm_bf16_wmma<<<dim3(BB / 128, HH / 64), 128, 0, stream>>>(
        AbigB, WbigB, nullptr, edgeF, edgeB, BB, HH, 4 * DD, /*flags=*/4);
    // tmp = edge @ We^T
    gemm_bf16_wmma<<<dim3(BB / 128, HH / 64), 128, 0, stream>>>(
        edgeB, WeB, nullptr, tmpF, nullptr, BB, HH, HH, /*flags=*/0);
    max_inplace<<<cdiv(nBH, 256), 256, 0, stream>>>(stF, tmpF, nBH);
  }

  // qt = relu(edge_4 + st_4)  (in place over edgeF)
  relu_add_inplace<<<cdiv(nBH, 256), 256, 0, stream>>>(edgeF, stF, nBH);

  // ---- query construction ----
  gather_idx<<<cdiv((size_t)BB * DD, 256), 256, 0, stream>>>(encB, idx, AbigB);
  // lin = enc_idx @ Wi^T + bi   -> tmpF
  gemm_bf16_wmma<<<dim3(BB / 128, HH / 64), 128, 0, stream>>>(
      AbigB, WiB, bi, tmpF, nullptr, BB, HH, DD, /*flags=*/1);
  // query = relu(relu(qt + lin) + lin) -> bf16 (reuse edgeB)
  build_query<<<cdiv(nBH, 256), 256, 0, stream>>>(edgeF, tmpF, edgeB, nBH);
  // inp = query @ Wq^T + bq  -> reuse stF as inp
  gemm_bf16_wmma<<<dim3(BB / 128, HH / 64), 128, 0, stream>>>(
      edgeB, WqB, bq, stF, nullptr, BB, HH, HH, /*flags=*/1);

  // ---- fused 4-way tanh attention: att[B, 4L] ----
  attn_fused<<<dim3((BB * LL) / 128, 4), 128, 0, stream>>>(encB, WcB, bc, V4, stF, mask, attF);

  // ---- softmax over batch (axis=0), then per-row categorical + outputs ----
  col_softmax<<<4 * LL, 256, 0, stream>>>(attF, BB, 4 * LL);
  sample_write<<<cdiv(BB, 256), 256, 0, stream>>>(attF, mask, (float*)d_out, BB);
}
